// GAT_47467978555679
// MI455X (gfx1250) — matrix operand order repaired and checked by reading, not compiled, not run on hardware
//
#include <hip/hip_runtime.h>

// ---- problem constants (match reference) ----
#define NN    100000   // nodes
#define NF    256      // input features
#define NHID  128      // heads*hid = 8*16
#define NHEAD 8
#define HID   16
#define NC    40       // classes
#define NEG_SLOPE 0.2f

typedef __attribute__((ext_vector_type(16))) __bf16 v16bf;
typedef __attribute__((ext_vector_type(8)))  __bf16 v8bf;
typedef __attribute__((ext_vector_type(8)))  float  v8f;
typedef __attribute__((ext_vector_type(4)))  unsigned v4u;
typedef __attribute__((ext_vector_type(8)))  int   v8i;
typedef __attribute__((ext_vector_type(4)))  int   v4i;

// ---------- helpers ----------
__device__ __forceinline__ float leaky(float x) {
  return x > 0.0f ? x : NEG_SLOPE * x;
}
// order-preserving float->uint encoding for atomicMax on floats
__device__ __forceinline__ unsigned f2ord(float f) {
  unsigned u = __float_as_uint(f);
  return (u & 0x80000000u) ? ~u : (u | 0x80000000u);
}
__device__ __forceinline__ float ord2f(unsigned u) {
  return (u & 0x80000000u) ? __uint_as_float(u ^ 0x80000000u) : __uint_as_float(~u);
}
#define ORD_NEG_INF 0x007FFFFFu   // f2ord(-inf)

// Two opaque ds_load_b128 from LDS (generic addr low 32 bits == LDS byte address).
// Inline asm keeps the loads from being folded to undef (TDM writes LDS behind
// the compiler's back) and guarantees the ds path + dscnt wait.
__device__ __forceinline__ void ds_load2_v8bf(const __bf16* p0, const __bf16* p1,
                                              v8bf& r0, v8bf& r1) {
  unsigned a0 = (unsigned)(unsigned long long)p0;
  unsigned a1 = (unsigned)(unsigned long long)p1;
  asm volatile("ds_load_b128 %0, %2\n\t"
               "ds_load_b128 %1, %3\n\t"
               "s_wait_dscnt 0x0"
               : "=&v"(r0), "=&v"(r1)
               : "v"(a0), "v"(a1));
}

// Issue one TDM 2-D tile load (bf16 elements) global->LDS.
//   ga        : global byte address of tile start
//   lds_off   : LDS byte offset of destination
//   w         : tile width in elements (row length, = tensor_dim0 & stride)
//   rows_tile : tile rows to move
//   rows_ok   : valid rows (tensor_dim1; OOB rows read as zero)
//   pad_iv    : pad_interval code (insert LDS pad every 2<<pad_iv DWORDs)
// pad_amount fixed at code 3 (= 4 DWORDs = 16 bytes) to stagger LDS banks.
__device__ __forceinline__ void tdm_load_2d_bf16(unsigned long long ga, unsigned lds_off,
                                                 int w, int rows_tile, int rows_ok,
                                                 int pad_iv) {
  v4u g0 = { 1u,                                   // count=1 (user descriptor)
             lds_off,                              // lds_addr
             (unsigned)(ga & 0xFFFFFFFFu),         // global_addr[31:0]
             (unsigned)((ga >> 32) & 0x01FFFFFFu)  // global_addr[56:32]
               | 0x80000000u };                    // type=2 ("image")
  int w0 = (1 << 16)            // data_size = 1 -> 2 bytes
         | (1 << 20)            // pad_enable
         | (pad_iv << 22)       // pad_interval
         | (3 << 25);           // pad_amount = 4 DWORDs (16B)
  v8i g1 = { w0,
             (int)((unsigned)w << 16),          // tensor_dim0[15:0] @ bits 63:48
             (int)((unsigned)rows_ok << 16),    // tensor_dim1[15:0] @ bits 95:80
             (int)((unsigned)w << 16),          // tile_dim0 @ bits 127:112 (dim1 hi = 0)
             rows_tile,                         // tile_dim1 @ bits 143:128
             w,                                 // tensor_dim0_stride[31:0]
             0, 0 };
  v4i z4 = { 0, 0, 0, 0 };
  v8i z8 = { 0, 0, 0, 0, 0, 0, 0, 0 };
  // 6-arg toolchain form: (g0, g1, g2, g3, extra, cpol)
  __builtin_amdgcn_tensor_load_to_lds(g0, g1, z4, z4, z8, 0);
}

// ---------- fills ----------
__global__ void fill_f32(float* p, float v, int n) {
  int i = blockIdx.x * blockDim.x + threadIdx.x;
  if (i < n) p[i] = v;
}
__global__ void fill_u32(unsigned* p, unsigned v, int n) {
  int i = blockIdx.x * blockDim.x + threadIdx.x;
  if (i < n) p[i] = v;
}

// ---------- precision / layout prep ----------
__global__ void conv_x_kernel(const float* __restrict__ x, __bf16* __restrict__ xb, int n) {
  int i = blockIdx.x * blockDim.x + threadIdx.x;
  if (i < n) xb[i] = (__bf16)x[i];
}
// Swizzle W1 [256,128] into WMMA-B tile layout: idx = ((kc*8+nb)*32+lane)*16+j
__global__ void swz_w1_kernel(const float* __restrict__ W1, __bf16* __restrict__ w1s) {
  int idx = blockIdx.x * blockDim.x + threadIdx.x;
  if (idx >= 8 * 8 * 32 * 16) return;
  int j  = idx & 15;
  int l  = (idx >> 4) & 31;
  int nb = (idx >> 9) & 7;
  int kc = (idx >> 12) & 7;
  int K = kc * 32 + ((l >> 4) << 3) + (j & 7) + ((j >> 3) << 4);   // lanes 0-15: K{0..7,16..23}, lanes 16-31: K{8..15,24..31}
  int N = nb * 16 + (l & 15);
  w1s[idx] = (__bf16)W1[K * NHID + N];
}
// Swizzle W2 [128,40] padded to N=48: idx = ((kc*3+nb)*32+lane)*16+j
__global__ void swz_w2_kernel(const float* __restrict__ W2, __bf16* __restrict__ w2s) {
  int idx = blockIdx.x * blockDim.x + threadIdx.x;
  if (idx >= 4 * 3 * 32 * 16) return;
  int j = idx & 15;
  int l = (idx >> 4) & 31;
  int t = idx >> 9;
  int nb = t % 3;
  int kc = t / 3;
  int K = kc * 32 + ((l >> 4) << 3) + (j & 7) + ((j >> 3) << 4);
  int N = nb * 16 + (l & 15);
  w2s[idx] = (N < NC) ? (__bf16)W2[K * NC + N] : (__bf16)0.0f;
}

// ---------- GEMM1: h1[100000,128] = xb[100000,256] @ W1 (bf16 in, f32 acc) ----------
// One block = one 16-row M-tile; TDM stages the A tile in LDS once, 8 waves = 8 N-tiles.
#define A1_STRIDE 264   // 256 elems + 8 elem (16B) TDM pad per row -> conflict-free banks
__global__ __launch_bounds__(256) void gemm1_kernel(const __bf16* __restrict__ xb,
                                                    const __bf16* __restrict__ w1s,
                                                    float* __restrict__ h1) {
  __shared__ __bf16 tileA[16 * A1_STRIDE];        // ~8.3 KB
  int wave = threadIdx.x >> 5;          // N-tile 0..7
  int lane = threadIdx.x & 31;
  int row0 = blockIdx.x * 16;           // M-tile base (6250 tiles exactly)

  if (wave == 0) {
    // rows: 512B = 128 DW per row -> pad_interval code 6 (128 DW)
    tdm_load_2d_bf16((unsigned long long)(const void*)(xb + (size_t)row0 * NF),
                     0u, NF, 16, 16, 6);
    __builtin_amdgcn_s_wait_tensorcnt(0);
  }
  __syncthreads();

  int kb = (lane >> 4) * 8;             // A layout: lanes 0-15 K{0..7,16..23}, 16-31 K{8..15,24..31}
  const __bf16* aRow = tileA + (lane & 15) * A1_STRIDE;
  v8f acc = {};
  #pragma unroll
  for (int kc = 0; kc < 8; ++kc) {
    v8bf alo, ahi;
    ds_load2_v8bf(aRow + kc * 32 + kb, aRow + kc * 32 + kb + 16, alo, ahi);
    v16bf a;
    #pragma unroll
    for (int j = 0; j < 8; ++j) { a[j] = alo[j]; a[8 + j] = ahi[j]; }
    v16bf b = *(const v16bf*)(w1s + ((size_t)((kc * 8 + wave) * 32 + lane) << 4));
    acc = __builtin_amdgcn_wmma_f32_16x16x32_bf16(false, a, false, b, (short)0, acc, false, false);
  }
  // C layout: lanes 0-15 -> M=v, lanes 16-31 -> M=8+v ; N = wave*16 + lane%16
  int n  = wave * 16 + (lane & 15);
  int mb = row0 + ((lane >> 4) << 3);
  #pragma unroll
  for (int v = 0; v < 8; ++v)
    h1[(size_t)(mb + v) * NHID + n] = acc[v];
}

// ---------- per-node attention logits, layer 1 ----------
__global__ void alpha1_kernel(const float* __restrict__ h1,
                              const float* __restrict__ a_src, const float* __restrict__ a_dst,
                              float* __restrict__ as1, float* __restrict__ ad1) {
  int idx = blockIdx.x * blockDim.x + threadIdx.x;   // n*8 + head
  if (idx >= NN * NHEAD) return;
  int n = idx >> 3, h = idx & 7;
  const float* hp = h1 + (size_t)n * NHID + h * HID;
  const float* sp = a_src + h * HID;
  const float* dp = a_dst + h * HID;
  float s = 0.f, d = 0.f;
  #pragma unroll
  for (int c = 0; c < HID; ++c) { s += hp[c] * sp[c]; d += hp[c] * dp[c]; }
  as1[idx] = s; ad1[idx] = d;
}

// ---------- edge pass 1a: segment max ----------
__global__ void edge_max1_kernel(const int* __restrict__ ei, int E,
                                 const float* __restrict__ as1, const float* __restrict__ ad1,
                                 unsigned* __restrict__ m1) {
  int e = blockIdx.x * blockDim.x + threadIdx.x;
  if (e >= E + NN) return;
  int s, d;
  if (e < E) { s = ei[e]; d = ei[E + e]; } else { s = d = e - E; }
  #pragma unroll
  for (int h = 0; h < NHEAD; ++h) {
    float lr = leaky(as1[s * NHEAD + h] + ad1[d * NHEAD + h]);
    atomicMax(&m1[d * NHEAD + h], f2ord(lr));
  }
}

// ---------- edge pass 1b: exp, denom, unnormalized weighted aggregation ----------
__global__ void edge_sum1_kernel(const int* __restrict__ ei, int E,
                                 const float* __restrict__ as1, const float* __restrict__ ad1,
                                 const unsigned* __restrict__ m1, const float* __restrict__ h1,
                                 float* __restrict__ s1, float* __restrict__ out1) {
  int idx = blockIdx.x * blockDim.x + threadIdx.x;   // (edge)*8 + head
  if (idx >= (E + NN) * NHEAD) return;
  int e = idx >> 3, h = idx & 7;
  int s, d;
  if (e < E) { s = ei[e]; d = ei[E + e]; } else { s = d = e - E; }
  float lr = leaky(as1[s * NHEAD + h] + ad1[d * NHEAD + h]);
  float ev = __expf(lr - ord2f(m1[d * NHEAD + h]));
  atomicAdd(&s1[d * NHEAD + h], ev);
  const float* hp = h1 + (size_t)s * NHID + h * HID;
  float* op = out1 + (size_t)d * NHID + h * HID;
  #pragma unroll
  for (int c = 0; c < HID; ++c) atomicAdd(&op[c], ev * hp[c]);
}

// ---------- normalize + bias + ELU + bf16 convert ----------
__global__ void finish1_kernel(const float* __restrict__ out1, const float* __restrict__ s1,
                               const float* __restrict__ b1, __bf16* __restrict__ h2b) {
  int idx = blockIdx.x * blockDim.x + threadIdx.x;   // n*128 + c
  if (idx >= NN * NHID) return;
  int n = idx >> 7, c = idx & 127;
  float v = out1[idx] / (s1[n * NHEAD + (c >> 4)] + 1e-16f) + b1[c];
  v = v > 0.0f ? v : (__expf(v) - 1.0f);             // ELU(alpha=1)
  h2b[idx] = (__bf16)v;
}

// ---------- GEMM2: g2[100000,40] = h2b[100000,128] @ W2 ----------
// One block = 8 M-tiles (128 rows); one TDM stages the whole 128x128 bf16 block tile.
#define A2_STRIDE 136   // 128 elems + 8 elem (16B) TDM pad per row
__global__ __launch_bounds__(256) void gemm2_kernel(const __bf16* __restrict__ h2b,
                                                    const __bf16* __restrict__ w2s,
                                                    float* __restrict__ g2) {
  __shared__ __bf16 tileA[128 * A2_STRIDE];          // ~34 KB
  int wave = threadIdx.x >> 5;
  int lane = threadIdx.x & 31;
  int row0b = blockIdx.x * 128;

  if (wave == 0) {
    int rows_ok = NN - row0b; if (rows_ok > 128) rows_ok = 128;  // OOB rows read as 0
    // rows: 256B = 64 DW per row -> pad_interval code 5 (64 DW)
    tdm_load_2d_bf16((unsigned long long)(const void*)(h2b + (size_t)row0b * NHID),
                     0u, NHID, 128, rows_ok, 5);
    __builtin_amdgcn_s_wait_tensorcnt(0);
  }
  __syncthreads();

  int row0 = row0b + wave * 16;                      // one M-tile per wave
  if (row0 >= NN) return;                            // uniform per wave -> EXEC stays full
  int kb = (lane >> 4) * 8;
  const __bf16* aRow = tileA + (wave * 16 + (lane & 15)) * A2_STRIDE;
  v8f acc0 = {}, acc1 = {}, acc2 = {};
  #pragma unroll
  for (int kc = 0; kc < 4; ++kc) {
    v8bf alo, ahi;
    ds_load2_v8bf(aRow + kc * 32 + kb, aRow + kc * 32 + kb + 16, alo, ahi);
    v16bf a;
    #pragma unroll
    for (int j = 0; j < 8; ++j) { a[j] = alo[j]; a[8 + j] = ahi[j]; }
    v16bf b0 = *(const v16bf*)(w2s + ((size_t)((kc * 3 + 0) * 32 + lane) << 4));
    v16bf b1 = *(const v16bf*)(w2s + ((size_t)((kc * 3 + 1) * 32 + lane) << 4));
    v16bf b2 = *(const v16bf*)(w2s + ((size_t)((kc * 3 + 2) * 32 + lane) << 4));
    acc0 = __builtin_amdgcn_wmma_f32_16x16x32_bf16(false, a, false, b0, (short)0, acc0, false, false);
    acc1 = __builtin_amdgcn_wmma_f32_16x16x32_bf16(false, a, false, b1, (short)0, acc1, false, false);
    acc2 = __builtin_amdgcn_wmma_f32_16x16x32_bf16(false, a, false, b2, (short)0, acc2, false, false);
  }
  int n0 = lane & 15;
  int mb = row0 + ((lane >> 4) << 3);
  #pragma unroll
  for (int v = 0; v < 8; ++v) {
    float* r = g2 + (size_t)(mb + v) * NC;
    r[n0] = acc0[v];
    r[16 + n0] = acc1[v];
    if (n0 < 8) r[32 + n0] = acc2[v];                // N 32..39 valid only
  }
}

// ---------- layer-2 attention (single head, 40 channels) ----------
__global__ void alpha2_kernel(const float* __restrict__ g2,
                              const float* __restrict__ a_src, const float* __restrict__ a_dst,
                              float* __restrict__ as2, float* __restrict__ ad2) {
  int n = blockIdx.x * blockDim.x + threadIdx.x;
  if (n >= NN) return;
  const float* gp = g2 + (size_t)n * NC;
  float s = 0.f, d = 0.f;
  #pragma unroll
  for (int c = 0; c < NC; ++c) { s += gp[c] * a_src[c]; d += gp[c] * a_dst[c]; }
  as2[n] = s; ad2[n] = d;
}

__global__ void edge_max2_kernel(const int* __restrict__ ei, int E,
                                 const float* __restrict__ as2, const float* __restrict__ ad2,
                                 unsigned* __restrict__ m2) {
  int e = blockIdx.x * blockDim.x + threadIdx.x;
  if (e >= E + NN) return;
  int s, d;
  if (e < E) { s = ei[e]; d = ei[E + e]; } else { s = d = e - E; }
  atomicMax(&m2[d], f2ord(leaky(as2[s] + ad2[d])));
}

__global__ void edge_sum2_kernel(const int* __restrict__ ei, int E,
                                 const float* __restrict__ as2, const float* __restrict__ ad2,
                                 const unsigned* __restrict__ m2, const float* __restrict__ g2,
                                 float* __restrict__ s2, float* __restrict__ out) {
  int e = blockIdx.x * blockDim.x + threadIdx.x;
  if (e >= E + NN) return;
  int s, d;
  if (e < E) { s = ei[e]; d = ei[E + e]; } else { s = d = e - E; }
  float ev = __expf(leaky(as2[s] + ad2[d]) - ord2f(m2[d]));
  atomicAdd(&s2[d], ev);
  const float* gp = g2 + (size_t)s * NC;
  float* op = out + (size_t)d * NC;
  #pragma unroll
  for (int c = 0; c < NC; ++c) atomicAdd(&op[c], ev * gp[c]);
}

__global__ void finish2_kernel(const float* __restrict__ s2, const float* __restrict__ b2,
                               float* __restrict__ out) {
  int idx = blockIdx.x * blockDim.x + threadIdx.x;   // n*40 + c
  if (idx >= NN * NC) return;
  int n = idx / NC, c = idx % NC;
  out[idx] = out[idx] / (s2[n] + 1e-16f) + b2[c];    // heads=1 -> mean is identity
}

// ---------- launch ----------
extern "C" void kernel_launch(void* const* d_in, const int* in_sizes, int n_in,
                              void* d_out, int out_size, void* d_ws, size_t ws_size,
                              hipStream_t stream) {
  const float* x      = (const float*)d_in[0];
  const int*   ei     = (const int*)d_in[1];
  const float* W1     = (const float*)d_in[2];
  const float* a_src1 = (const float*)d_in[3];
  const float* a_dst1 = (const float*)d_in[4];
  const float* b1     = (const float*)d_in[5];
  const float* W2     = (const float*)d_in[6];
  const float* a_src2 = (const float*)d_in[7];
  const float* a_dst2 = (const float*)d_in[8];
  const float* b2     = (const float*)d_in[9];
  const int E = in_sizes[1] / 2;

  char* p = (char*)d_ws;
  auto carve = [&](size_t bytes) { char* r = p; p += (bytes + 255) & ~(size_t)255; return r; };
  __bf16*   xb   = (__bf16*)  carve((size_t)NN * NF * 2);
  __bf16*   w1s  = (__bf16*)  carve((size_t)8 * 8 * 32 * 16 * 2);
  __bf16*   w2s  = (__bf16*)  carve((size_t)4 * 3 * 32 * 16 * 2);
  float*    h1   = (float*)   carve((size_t)NN * NHID * 4);
  float*    as1  = (float*)   carve((size_t)NN * NHEAD * 4);
  float*    ad1  = (float*)   carve((size_t)NN * NHEAD * 4);
  unsigned* m1   = (unsigned*)carve((size_t)NN * NHEAD * 4);
  float*    s1   = (float*)   carve((size_t)NN * NHEAD * 4);
  float*    out1 = (float*)   carve((size_t)NN * NHID * 4);
  __bf16*   h2b  = (__bf16*)  carve((size_t)NN * NHID * 2);
  float*    g2   = (float*)   carve((size_t)NN * NC * 4);
  float*    as2  = (float*)   carve((size_t)NN * 4);
  float*    ad2  = (float*)   carve((size_t)NN * 4);
  unsigned* m2   = (unsigned*)carve((size_t)NN * 4);
  float*    s2   = (float*)   carve((size_t)NN * 4);
  float*    outf = (float*)d_out;

  const int B = 256;
  auto G = [](long long n, int b) { return (unsigned)((n + b - 1) / b); };

  // init accumulators (every call -> deterministic under graph replay)
  fill_u32<<<G(NN * NHEAD, B), B, 0, stream>>>(m1, ORD_NEG_INF, NN * NHEAD);
  fill_f32<<<G(NN * NHEAD, B), B, 0, stream>>>(s1, 0.f, NN * NHEAD);
  fill_f32<<<G((long long)NN * NHID, B), B, 0, stream>>>(out1, 0.f, NN * NHID);
  fill_u32<<<G(NN, B), B, 0, stream>>>(m2, ORD_NEG_INF, NN);
  fill_f32<<<G(NN, B), B, 0, stream>>>(s2, 0.f, NN);
  fill_f32<<<G((long long)NN * NC, B), B, 0, stream>>>(outf, 0.f, NN * NC);

  // precision / layout prep
  conv_x_kernel<<<G((long long)NN * NF, B), B, 0, stream>>>(x, xb, NN * NF);
  swz_w1_kernel<<<G(8 * 8 * 32 * 16, B), B, 0, stream>>>(W1, w1s);
  swz_w2_kernel<<<G(4 * 3 * 32 * 16, B), B, 0, stream>>>(W2, w2s);

  // layer 1
  gemm1_kernel<<<NN / 16, 256, 0, stream>>>(xb, w1s, h1);
  alpha1_kernel<<<G(NN * NHEAD, B), B, 0, stream>>>(h1, a_src1, a_dst1, as1, ad1);
  edge_max1_kernel<<<G(E + NN, B), B, 0, stream>>>(ei, E, as1, ad1, m1);
  edge_sum1_kernel<<<G((long long)(E + NN) * NHEAD, B), B, 0, stream>>>(ei, E, as1, ad1, m1, h1, s1, out1);
  finish1_kernel<<<G((long long)NN * NHID, B), B, 0, stream>>>(out1, s1, b1, h2b);

  // layer 2
  gemm2_kernel<<<G(NN, 128), 256, 0, stream>>>(h2b, w2s, g2);
  alpha2_kernel<<<G(NN, B), B, 0, stream>>>(g2, a_src2, a_dst2, as2, ad2);
  edge_max2_kernel<<<G(E + NN, B), B, 0, stream>>>(ei, E, as2, ad2, m2);
  edge_sum2_kernel<<<G(E + NN, B), B, 0, stream>>>(ei, E, as2, ad2, m2, g2, s2, outf);
  finish2_kernel<<<G((long long)NN * NC, B), B, 0, stream>>>(s2, b2, outf);
}
